// ShapeRetrieval_19585050869761
// MI455X (gfx1250) — matrix operand order repaired and checked by reading, not compiled, not run on hardware
//
#include <hip/hip_runtime.h>
#include <hip/hip_bf16.h>

typedef __attribute__((ext_vector_type(16))) _Float16 v16h;
typedef __attribute__((ext_vector_type(8)))  _Float16 v8h;
typedef __attribute__((ext_vector_type(4)))  _Float16 v4h;
typedef __attribute__((ext_vector_type(8)))  float    v8f;
typedef __attribute__((ext_vector_type(4)))  int      v4i;

typedef __attribute__((address_space(1))) v4i* as1_v4i_ptr;  // global int4*
typedef __attribute__((address_space(3))) v4i* as3_v4i_ptr;  // LDS int4*

#define D_DIM 512
#define TQ 128               // query rows per block
#define TB 128               // bank rows per block
#define KC 32                // K-chunk = one WMMA k-step
#define NC (D_DIM / KC)      // 16 chunks
#define LDH (KC + 8)         // padded LDS row stride in halves (80B, 16B-aligned)

// CDNA5 async global->LDS DMA (ASYNCcnt-tracked), guarded for portability.
#if defined(__has_builtin)
#if __has_builtin(__builtin_amdgcn_global_load_async_to_lds_b128) && \
    __has_builtin(__builtin_amdgcn_s_wait_asynccnt)
#define USE_ASYNC_LDS 1
#endif
#endif

#if defined(USE_ASYNC_LDS)
#define WAIT_ASYNC(n) __builtin_amdgcn_s_wait_asynccnt(n)
#else
#define WAIT_ASYNC(n)
#endif

static __device__ __forceinline__ void copy16_g2l(const _Float16* g, _Float16* l) {
#if defined(USE_ASYNC_LDS)
    __builtin_amdgcn_global_load_async_to_lds_b128(
        (as1_v4i_ptr)g, (as3_v4i_ptr)l, 0, 0);
#else
    *(uint4*)l = *(const uint4*)g;
#endif
}

static __device__ __forceinline__ v16h cat8(v8h lo, v8h hi) {
    return __builtin_shufflevector(lo, hi, 0,1,2,3,4,5,6,7,8,9,10,11,12,13,14,15);
}

// Monotonic u32 key: total order over floats via unsigned compare.
static __device__ __forceinline__ unsigned int fkey(float v) {
    unsigned int b = __float_as_uint(v);
    return (b & 0x80000000u) ? ~b : (b | 0x80000000u);
}

__global__ void init_best_kernel(unsigned long long* best, int n) {
    int i = blockIdx.x * blockDim.x + threadIdx.x;
    if (i < n) best[i] = 0ull;
}

// One row per 128-thread block: sum of squares -> rsqrt -> scaled f16 row.
__global__ void l2norm_f16_kernel(const float* __restrict__ src,
                                  _Float16* __restrict__ dst) {
    const int row = blockIdx.x;
    const int tid = threadIdx.x;              // 0..127
    const float4 v = ((const float4*)(src + (size_t)row * D_DIM))[tid];
    float ss = v.x*v.x + v.y*v.y + v.z*v.z + v.w*v.w;
    #pragma unroll
    for (int m = 16; m >= 1; m >>= 1) ss += __shfl_xor(ss, m, 32);
    __shared__ float part[4];
    if ((tid & 31) == 0) part[tid >> 5] = ss;
    __syncthreads();
    const float tot = part[0] + part[1] + part[2] + part[3];
    const float inv = 1.0f / fmaxf(sqrtf(tot), 1e-12f);
    v4h o;
    o.x = (_Float16)(v.x * inv);
    o.y = (_Float16)(v.y * inv);
    o.z = (_Float16)(v.z * inv);
    o.w = (_Float16)(v.w * inv);
    ((v4h*)(dst + (size_t)row * D_DIM))[tid] = o;
}

// A layout (ISA 16-bit A 16x32): lanes<16 K={0..7,16..23}, lanes>=16 +8.
static __device__ __forceinline__ v16h frag_a(const _Float16* base, int row, int ko) {
    const _Float16* p = base + row * LDH + ko;
    return cat8(*(const v8h*)p, *(const v8h*)(p + 16));
}
// B layout: lane holds one column (= bank row); lanes<16 K={0..7,16..23}, lanes>=16 +8.
static __device__ __forceinline__ v16h frag_b(const _Float16* base, int row, int ko) {
    const _Float16* p = base + row * LDH + ko;
    return cat8(*(const v8h*)p, *(const v8h*)(p + 16));
}

__global__ void __launch_bounds__(256)
gemm_argmax_kernel(const _Float16* __restrict__ qf,
                   const _Float16* __restrict__ bf,
                   unsigned long long* __restrict__ best,
                   int N, int M) {
    __shared__ _Float16 sQ[2][TQ * LDH];
    __shared__ _Float16 sB[2][TB * LDH];

    const int tid   = threadIdx.x;
    const int lane  = tid & 31;
    const int wave  = tid >> 5;         // 0..7, each owns 16 q rows
    const int hl    = lane >> 4;        // lane half (0/1)
    const int r16   = lane & 15;
    const int qbase = blockIdx.x * TQ;
    const int bbase = blockIdx.y * TB;

    v8f acc[8];
    #pragma unroll
    for (int s = 0; s < 8; ++s) acc[s] = (v8f)0.0f;

    // Stage one K-chunk (128 rows x 32 halves, both tiles): 4 x 16B per thread.
    // Out-of-range rows are clamped (branch-free); their WMMA columns/rows are
    // discarded by the n<M / qrow<N guards in the argmax, so duplicates are safe.
    auto copy_chunk = [&](int c, int buf) {
        const int kk = c * KC;
        #pragma unroll
        for (int t = 0; t < 2; ++t) {
            const int g    = tid + t * 256;     // 0..511
            const int row  = g >> 2;            // 0..127
            const int c8   = (g & 3) * 8;       // half offset
            const int qrow = min(qbase + row, N - 1);
            const int brow = min(bbase + row, M - 1);
            copy16_g2l(qf + (size_t)qrow * D_DIM + kk + c8, &sQ[buf][row * LDH + c8]);
            copy16_g2l(bf + (size_t)brow * D_DIM + kk + c8, &sB[buf][row * LDH + c8]);
        }
    };

    // 8 WMMAs per wave; B-fragment loads pipelined one subtile ahead.
    auto compute_chunk = [&](int buf) {
        const int ako = hl * 8;
        const int bko = hl * 8;
        const v16h a = frag_a(&sQ[buf][0], wave * 16 + r16, ako);
        v16h bcur = frag_b(&sB[buf][0], r16, bko);
        #pragma unroll
        for (int sub = 0; sub < 8; ++sub) {
            v16h bnext = bcur;
            if (sub < 7) bnext = frag_b(&sB[buf][0], (sub + 1) * 16 + r16, bko);
            acc[sub] = __builtin_amdgcn_wmma_f32_16x16x32_f16(
                false, a, false, bcur, (short)0, acc[sub], false, false);
            bcur = bnext;
        }
    };

    copy_chunk(0, 0);
    for (int c = 0; c < NC - 1; ++c) {
        copy_chunk(c + 1, (c + 1) & 1);   // prefetch next chunk into other buffer
        WAIT_ASYNC(4);                    // <=4 outstanding: only the prefetch remains
        __syncthreads();                  // all threads' chunk c visible in LDS
        compute_chunk(c & 1);
        __syncthreads();                  // done reading before buffer is overwritten
    }
    WAIT_ASYNC(0);
    __syncthreads();
    compute_chunk((NC - 1) & 1);

    // Per-row argmax. C/D layout: VGPR r, lanes<16 -> M=r, lanes>=16 -> M=r+8; N=lane%16.
    #pragma unroll
    for (int r = 0; r < 8; ++r) {
        const int qrow = qbase + wave * 16 + hl * 8 + r;
        unsigned long long bp = 0ull;
        #pragma unroll
        for (int sub = 0; sub < 8; ++sub) {
            const int n = bbase + sub * 16 + r16;
            const unsigned long long p =
                ((unsigned long long)fkey(acc[sub][r]) << 32) |
                (unsigned long long)(unsigned int)(~(unsigned int)n);
            if (n < M && p > bp) bp = p;
        }
        #pragma unroll
        for (int m2 = 8; m2 >= 1; m2 >>= 1) {
            unsigned long long o = __shfl_xor(bp, m2, 32);
            if (o > bp) bp = o;
        }
        if (r16 == 0 && qrow < N)
            atomicMax(best + qrow, bp);
    }
}

__global__ void finalize_kernel(const unsigned long long* __restrict__ best,
                                const int* __restrict__ cat,
                                const int* __restrict__ shp,
                                float* __restrict__ out, int N, int M) {
    int q = blockIdx.x * blockDim.x + threadIdx.x;
    if (q >= N) return;
    unsigned int low = (unsigned int)(best[q] & 0xFFFFFFFFull);
    unsigned int n = ~low;
    if (n >= (unsigned int)M) n = 0;
    out[q]     = (float)cat[n];
    out[N + q] = (float)shp[n];
}

extern "C" void kernel_launch(void* const* d_in, const int* in_sizes, int n_in,
                              void* d_out, int out_size, void* d_ws, size_t ws_size,
                              hipStream_t stream) {
    (void)n_in; (void)out_size; (void)ws_size;
    const int N = in_sizes[0] / D_DIM;   // 4096 queries
    const int M = in_sizes[1] / D_DIM;   // 100000 bank rows

    const float* q  = (const float*)d_in[0];
    const float* b  = (const float*)d_in[1];
    const int* cat  = (const int*)d_in[2];
    const int* shp  = (const int*)d_in[3];
    float* out      = (float*)d_out;

    // ws layout: [best u64 x N][q f16 NxD][bank f16 MxD]  (~107 MB total)
    unsigned long long* best = (unsigned long long*)d_ws;
    size_t off = (((size_t)N * 8) + 255) & ~(size_t)255;
    _Float16* qf = (_Float16*)((char*)d_ws + off);
    _Float16* bf = qf + (size_t)N * D_DIM;

    init_best_kernel<<<(N + 255) / 256, 256, 0, stream>>>(best, N);
    l2norm_f16_kernel<<<N, 128, 0, stream>>>(q, qf);
    l2norm_f16_kernel<<<M, 128, 0, stream>>>(b, bf);

    dim3 grid((N + TQ - 1) / TQ, (M + TB - 1) / TB);  // x=q-tiles adjacent -> L2 reuse of bank tile
    gemm_argmax_kernel<<<grid, 256, 0, stream>>>(qf, bf, best, N, M);

    finalize_kernel<<<(N + 255) / 256, 256, 0, stream>>>(best, cat, shp, out, N, M);
}
